// MOE_MLP_15247133900829
// MI455X (gfx1250) — matrix operand order repaired and checked by reading, not compiled, not run on hardware
//
#include <hip/hip_runtime.h>
#include <math.h>

// ---------------- types ----------------
typedef __bf16 bf16_t;
typedef __attribute__((ext_vector_type(4)))  bf16_t v4bf;
typedef __attribute__((ext_vector_type(8)))  bf16_t v8bf;
typedef __attribute__((ext_vector_type(16))) bf16_t v16bf;
typedef __attribute__((ext_vector_type(8)))  float  v8f;

// ---------------- problem constants ----------------
#define T_TOK   4096          // 2*2048 tokens
#define DMODEL  1024
#define HDIM    1024
#define NEXP    32
#define TOPK    2
#define MT      64            // M tile (rows of routed tokens) per block
#define NT      128           // N tile per block
#define MAXROWS (T_TOK*TOPK + NEXP*MT)   // 8192 + 2048 = 10240 padded routed rows

// ===================================================================
// 0) zero the output (harness poisons it; we accumulate with atomics)
// ===================================================================
__global__ __launch_bounds__(256) void zero_out_kernel(float* __restrict__ out, int n)
{
    int i4 = (blockIdx.x * 256 + threadIdx.x) * 4;
    if (i4 + 3 < n) {
        *(float4*)(out + i4) = make_float4(0.f, 0.f, 0.f, 0.f);
    } else {
        for (int i = i4; i < n; ++i) out[i] = 0.f;
    }
}

// ===================================================================
// 1) init routing scratch (re-done every launch: no cross-call state)
// ===================================================================
__global__ __launch_bounds__(256) void route_init_kernel(int* __restrict__ rowtok,
                                                         int* __restrict__ counts,
                                                         int* __restrict__ cursor)
{
    int i = blockIdx.x * 256 + threadIdx.x;
    if (i < MAXROWS) rowtok[i] = -1;
    if (i < NEXP) { counts[i] = 0; cursor[i] = 0; }
}

// ===================================================================
// 2) gating: one wave32 per token; lane == expert
//    scores = softmax(x @ Wg); top2; weights = softmax(top2 scores)
// ===================================================================
__global__ __launch_bounds__(128) void gate_kernel(const float* __restrict__ x,
                                                   const float* __restrict__ Wg,
                                                   int*   __restrict__ eidx,
                                                   float* __restrict__ wgt,
                                                   int*   __restrict__ counts)
{
    const int lane = threadIdx.x & 31;
    const int wsub = threadIdx.x >> 5;
    const int t    = blockIdx.x * 4 + wsub;

    // dot(x[t], Wg[:,lane]) : broadcast x chunks through shuffles
    float acc = 0.f;
    const float* xr = x + (size_t)t * DMODEL;
    for (int i0 = 0; i0 < DMODEL; i0 += 32) {
        float xv = xr[i0 + lane];
        #pragma unroll
        for (int j = 0; j < 32; ++j) {
            float xj = __shfl(xv, j, 32);
            acc = fmaf(xj, Wg[(i0 + j) * NEXP + lane], acc);
        }
    }
    // softmax over 32 lanes
    float m = acc;
    for (int off = 16; off > 0; off >>= 1) m = fmaxf(m, __shfl_xor(m, off, 32));
    float p = __expf(acc - m);
    float s = p;
    for (int off = 16; off > 0; off >>= 1) s += __shfl_xor(s, off, 32);
    p /= s;

    // top-1 (ties -> lowest index, matching jax.lax.top_k stability)
    float v0 = p; int i0e = lane;
    for (int off = 16; off > 0; off >>= 1) {
        float ov = __shfl_xor(v0, off, 32); int oi = __shfl_xor(i0e, off, 32);
        if (ov > v0 || (ov == v0 && oi < i0e)) { v0 = ov; i0e = oi; }
    }
    // top-2
    float p2 = (lane == i0e) ? -1e30f : p;
    float v1 = p2; int i1e = lane;
    for (int off = 16; off > 0; off >>= 1) {
        float ov = __shfl_xor(v1, off, 32); int oi = __shfl_xor(i1e, off, 32);
        if (ov > v1 || (ov == v1 && oi < i1e)) { v1 = ov; i1e = oi; }
    }
    // renormalize the two probabilities with another softmax (v0 >= v1)
    float e1 = __expf(v1 - v0);
    float w0 = 1.f / (1.f + e1);
    float w1 = e1 / (1.f + e1);

    if (lane == 0) {
        eidx[t * 2 + 0] = i0e;  wgt[t * 2 + 0] = w0;
        eidx[t * 2 + 1] = i1e;  wgt[t * 2 + 1] = w1;
        atomicAdd(&counts[i0e], 1);
        atomicAdd(&counts[i1e], 1);
    }
}

// ===================================================================
// 3) per-expert padded offsets (pad each bucket to a multiple of MT)
// ===================================================================
__global__ void offsets_kernel(const int* __restrict__ counts, int* __restrict__ poff)
{
    if (threadIdx.x == 0) {
        int off = 0;
        for (int e = 0; e < NEXP; ++e) {
            poff[e] = off;
            off += ((counts[e] + MT - 1) / MT) * MT;
        }
        poff[NEXP] = off;
    }
}

// ===================================================================
// 4) scatter routed (token, k) pairs into expert buckets
// ===================================================================
__global__ __launch_bounds__(256) void scatter_kernel(const int* __restrict__ eidx,
                                                      const float* __restrict__ wgt,
                                                      const int* __restrict__ poff,
                                                      int*   __restrict__ cursor,
                                                      int*   __restrict__ rowtok,
                                                      float* __restrict__ roww)
{
    int i = blockIdx.x * 256 + threadIdx.x;
    if (i >= T_TOK * TOPK) return;
    int e = eidx[i];
    int pos = poff[e] + atomicAdd(&cursor[e], 1);
    rowtok[pos] = i >> 1;
    roww[pos]   = wgt[i];
}

// ===================================================================
// 5) x -> bf16
// ===================================================================
__global__ __launch_bounds__(256) void xconv_kernel(const float* __restrict__ x,
                                                    bf16_t* __restrict__ xbf)
{
    int i4 = (blockIdx.x * 256 + threadIdx.x) * 4;
    float4 f = *(const float4*)(x + i4);
    v4bf o;
    o[0] = (bf16_t)f.x; o[1] = (bf16_t)f.y; o[2] = (bf16_t)f.z; o[3] = (bf16_t)f.w;
    *(v4bf*)(xbf + i4) = o;
}

// ===================================================================
// software-pipelined 16x64 MMA tile over K (K % 64 == 0)
//   A: bf16 rows (lane = M row, kseg folded into aptr by caller)
//   B: fp32 [K][ldb] row-major; lane = N column (k strided, table order); converted to bf16 in regs
// ===================================================================
__device__ __forceinline__ void mma_tile_16x64(const bf16_t* __restrict__ aptr,
                                               const float*  __restrict__ bbase,
                                               int ldb, int wn0, int lane, int K,
                                               v8f acc[4])
{
    const float* brow0 = bbase + (size_t)((lane >> 4) * 8) * ldb + wn0 + (lane & 15);

    float4 raw0[16], raw1[16];
    v8bf al0, ah0, al1, ah1;

    auto issue = [&](float4 (&raw)[16], v8bf& alo, v8bf& ahi, int k0) {
        alo = *(const v8bf*)(aptr + k0);
        ahi = *(const v8bf*)(aptr + k0 + 16);
        const float* br = brow0 + (size_t)k0 * ldb;
        __builtin_prefetch(br + (size_t)64 * ldb, 0, 1);   // warm L2 two steps ahead
        #pragma unroll
        for (int j = 0; j < 4; ++j) {
            const float* bp = br + j * 16;
            #pragma unroll
            for (int t = 0; t < 4; ++t) {
                const int kb = t * 4 + (t >= 2 ? 8 : 0);
                raw[j * 4 + t] = make_float4(bp[(size_t)kb * ldb], bp[(size_t)(kb + 1) * ldb],
                                             bp[(size_t)(kb + 2) * ldb], bp[(size_t)(kb + 3) * ldb]);
            }
        }
    };

    auto compute = [&](float4 (&raw)[16], v8bf& alo, v8bf& ahi) {
        v16bf a = __builtin_shufflevector(alo, ahi,
                      0,1,2,3,4,5,6,7,8,9,10,11,12,13,14,15);
        #pragma unroll
        for (int j = 0; j < 4; ++j) {
            v16bf b;
            #pragma unroll
            for (int t = 0; t < 4; ++t) {
                float4 f = raw[j * 4 + t];
                b[t * 4 + 0] = (bf16_t)f.x;
                b[t * 4 + 1] = (bf16_t)f.y;
                b[t * 4 + 2] = (bf16_t)f.z;
                b[t * 4 + 3] = (bf16_t)f.w;
            }
            acc[j] = __builtin_amdgcn_wmma_f32_16x16x32_bf16(
                         false, a, false, b, (short)0, acc[j], false, false);
        }
    };

    issue(raw0, al0, ah0, 0);                       // prime the pipeline
    for (int k0 = 0; k0 < K; k0 += 64) {
        issue(raw1, al1, ah1, k0 + 32);             // loads for k0+32 in flight
        compute(raw0, al0, ah0);                    // WMMA on k0
        if (k0 + 64 < K) issue(raw0, al0, ah0, k0 + 64);
        compute(raw1, al1, ah1);                    // WMMA on k0+32
    }
}

// ===================================================================
// 6) GEMM-H: H = gelu(Xg @ W1[e] + b1[e]) -> hbuf (bf16)
//    block = 8 waves: wave (msub = w&3, nhalf = w>>2) computes 16x64
// ===================================================================
__global__ __launch_bounds__(256) void gemm_h_kernel(const bf16_t* __restrict__ xbf,
                                                     const float*  __restrict__ W1,
                                                     const float*  __restrict__ b1,
                                                     const int*    __restrict__ rowtok,
                                                     const int*    __restrict__ poff,
                                                     const int*    __restrict__ counts,
                                                     bf16_t*       __restrict__ hbuf)
{
    const int row_base = blockIdx.x * MT;
    if (row_base >= poff[NEXP]) return;
    int e = 0;
    while (poff[e + 1] <= row_base) ++e;
    const int limit = poff[e] + counts[e];

    const int lane  = threadIdx.x & 31;
    const int wave  = threadIdx.x >> 5;
    const int msub  = wave & 3;
    const int nhalf = wave >> 2;
    const int wn0   = blockIdx.y * NT + nhalf * 64;

    const int arow = row_base + msub * 16 + (lane & 15);
    int tok = rowtok[arow]; if (tok < 0) tok = 0;
    const bf16_t* aptr  = xbf + (size_t)tok * DMODEL + (lane >> 4) * 8;
    const float*  bbase = W1 + (size_t)e * DMODEL * HDIM;

    v8f acc[4];
    #pragma unroll
    for (int j = 0; j < 4; ++j)
        #pragma unroll
        for (int r = 0; r < 8; ++r) acc[j][r] = 0.f;

    mma_tile_16x64(aptr, bbase, HDIM, wn0, lane, DMODEL, acc);

    const int colbase = wn0 + (lane & 15);
    const int hh = lane >> 4;
    #pragma unroll
    for (int j = 0; j < 4; ++j) {
        const int col = colbase + j * 16;
        const float bias = b1[e * HDIM + col];
        #pragma unroll
        for (int r = 0; r < 8; ++r) {
            const int grow = row_base + msub * 16 + hh * 8 + r;
            if (grow < limit) {
                float v = acc[j][r] + bias;
                v = 0.5f * v * (1.0f + erff(v * 0.70710678118654752f));  // exact gelu
                hbuf[(size_t)grow * HDIM + col] = (bf16_t)v;
            }
        }
    }
}

// ===================================================================
// 7) GEMM-Y: y[tok] += w_row * (H @ W2[e] + b2[e])   (fp32 atomics)
// ===================================================================
__global__ __launch_bounds__(256) void gemm_y_kernel(const bf16_t* __restrict__ hbuf,
                                                     const float*  __restrict__ W2,
                                                     const float*  __restrict__ b2,
                                                     const int*    __restrict__ rowtok,
                                                     const float*  __restrict__ roww,
                                                     const int*    __restrict__ poff,
                                                     const int*    __restrict__ counts,
                                                     float*        __restrict__ out)
{
    const int row_base = blockIdx.x * MT;
    if (row_base >= poff[NEXP]) return;
    int e = 0;
    while (poff[e + 1] <= row_base) ++e;
    const int limit = poff[e] + counts[e];

    const int lane  = threadIdx.x & 31;
    const int wave  = threadIdx.x >> 5;
    const int msub  = wave & 3;
    const int nhalf = wave >> 2;
    const int wn0   = blockIdx.y * NT + nhalf * 64;

    const int arow = row_base + msub * 16 + (lane & 15);
    const bf16_t* aptr  = hbuf + (size_t)arow * HDIM + (lane >> 4) * 8;
    const float*  bbase = W2 + (size_t)e * HDIM * DMODEL;

    v8f acc[4];
    #pragma unroll
    for (int j = 0; j < 4; ++j)
        #pragma unroll
        for (int r = 0; r < 8; ++r) acc[j][r] = 0.f;

    mma_tile_16x64(aptr, bbase, DMODEL, wn0, lane, HDIM, acc);

    const int colbase = wn0 + (lane & 15);
    const int hh = lane >> 4;
    float bias[4];
    #pragma unroll
    for (int j = 0; j < 4; ++j) bias[j] = b2[e * DMODEL + colbase + j * 16];

    #pragma unroll
    for (int r = 0; r < 8; ++r) {
        const int grow = row_base + msub * 16 + hh * 8 + r;
        if (grow < limit) {
            const int   tok = rowtok[grow];
            const float w   = roww[grow];
            #pragma unroll
            for (int j = 0; j < 4; ++j) {
                const float v = (acc[j][r] + bias[j]) * w;
                atomicAdd(out + (size_t)tok * DMODEL + colbase + j * 16, v);
            }
        }
    }
}

// ===================================================================
// launcher
// ===================================================================
extern "C" void kernel_launch(void* const* d_in, const int* in_sizes, int n_in,
                              void* d_out, int out_size, void* d_ws, size_t ws_size,
                              hipStream_t stream)
{
    const float* x  = (const float*)d_in[0];
    const float* Wg = (const float*)d_in[1];
    const float* W1 = (const float*)d_in[2];
    const float* b1 = (const float*)d_in[3];
    const float* W2 = (const float*)d_in[4];
    const float* b2 = (const float*)d_in[5];
    float* out = (float*)d_out;

    // workspace carve-up (~29.5 MB total)
    char* ws = (char*)d_ws;
    size_t off = 0;
    auto carve = [&](size_t bytes) -> void* {
        void* p = ws + off;
        off = (off + bytes + 255) & ~(size_t)255;
        return p;
    };
    bf16_t* xbf    = (bf16_t*)carve((size_t)T_TOK * DMODEL * sizeof(bf16_t));
    bf16_t* hbuf   = (bf16_t*)carve((size_t)MAXROWS * HDIM * sizeof(bf16_t));
    int*    eidx   = (int*)   carve((size_t)T_TOK * TOPK * sizeof(int));
    float*  wgt    = (float*) carve((size_t)T_TOK * TOPK * sizeof(float));
    int*    rowtok = (int*)   carve((size_t)MAXROWS * sizeof(int));
    float*  roww   = (float*) carve((size_t)MAXROWS * sizeof(float));
    int*    counts = (int*)   carve(NEXP * sizeof(int));
    int*    cursor = (int*)   carve(NEXP * sizeof(int));
    int*    poff   = (int*)   carve((NEXP + 1) * sizeof(int));
    (void)ws_size; (void)in_sizes; (void)n_in;

    zero_out_kernel<<<(out_size + 1023) / 1024, 256, 0, stream>>>(out, out_size);
    route_init_kernel<<<(MAXROWS + 255) / 256, 256, 0, stream>>>(rowtok, counts, cursor);
    gate_kernel<<<T_TOK / 4, 128, 0, stream>>>(x, Wg, eidx, wgt, counts);
    offsets_kernel<<<1, 32, 0, stream>>>(counts, poff);
    scatter_kernel<<<(T_TOK * TOPK + 255) / 256, 256, 0, stream>>>(eidx, wgt, poff, cursor, rowtok, roww);
    xconv_kernel<<<(T_TOK * DMODEL / 4 + 255) / 256, 256, 0, stream>>>(x, xbf);

    dim3 gh(MAXROWS / MT, HDIM / NT);
    gemm_h_kernel<<<gh, 256, 0, stream>>>(xbf, W1, b1, rowtok, poff, counts, hbuf);
    dim3 gy(MAXROWS / MT, DMODEL / NT);
    gemm_y_kernel<<<gy, 256, 0, stream>>>(hbuf, W2, b2, rowtok, roww, poff, counts, out);
}
